// SchNetInteractionAngularAtom_2774548873990
// MI455X (gfx1250) — matrix operand order repaired and checked by reading, compiled, not run on hardware
//
#include <hip/hip_runtime.h>
#include <hip/hip_bf16.h>
#include <stdint.h>

// ---------------------------------------------------------------------------
// Types for gfx1250 WMMA (wave32): V_WMMA_F32_16X16X32_BF16
// ---------------------------------------------------------------------------
typedef __bf16 bf16;
typedef __attribute__((ext_vector_type(16))) __bf16      v16bf;
typedef __attribute__((ext_vector_type(8)))  float       v8f;
typedef __attribute__((ext_vector_type(8)))  unsigned    v8u;

#define LN2F 0.6931471805599453f

__device__ __forceinline__ float ssp(float x) {
    // shifted softplus: log(0.5*exp(x)+0.5) = softplus(x) - ln2, stable form
    return fmaxf(x, 0.f) + __logf(1.f + __expf(-fabsf(x))) - LN2F;
}

__device__ __forceinline__ unsigned pack2(float lo, float hi) {
    union { bf16 h[2]; unsigned u; } c;
    c.h[0] = (bf16)lo; c.h[1] = (bf16)hi;
    return c.u;
}

__device__ __forceinline__ v8f wmma_bf16(v16bf a, v16bf b, v8f c) {
    // 8 args: (neg_a, A, neg_b, B, c_mod, C, reuse_a, reuse_b)
    return __builtin_amdgcn_wmma_f32_16x16x32_bf16(false, a, false, b,
                                                   (short)0, c, false, false);
}

// A-matrix 16x32 bf16 fragment from a K-pair-packed u32 array (row-major,
// row_stride in u32). ISA layout: lane m=L%16, g=L/16; VGPR v0..3 hold
// K = g*8 + 2v,2v+1 ; v4..7 hold K = 16 + g*8 + 2(v-4),+1.
// v0..3 and v4..7 are each 4 contiguous u32 -> two ds_load_b128.
__device__ __forceinline__ v16bf load_afrag(const unsigned* base, int row_stride,
                                            int rbase, int kpair_base, int lane) {
    const int m = lane & 15, g = lane >> 4;
    const unsigned* rp = base + (size_t)(rbase + m) * row_stride + kpair_base;
    v8u au;
#pragma unroll
    for (int v = 0; v < 4; ++v) au[v]     = rp[g * 4 + v];
#pragma unroll
    for (int v = 0; v < 4; ++v) au[4 + v] = rp[8 + g * 4 + v];
    return __builtin_bit_cast(v16bf, au);
}

// B-matrix 32x16 bf16 fragment from a *fragment-order swizzled* weight buffer:
// Wf[ct][ks][lane][v] with v = 8 contiguous u32 per lane -> 2 global_load_b128.
__device__ __forceinline__ v16bf load_bfrag_sw(const unsigned* Wf, int KS,
                                               int ks, int ct, int lane) {
    const unsigned* p = Wf + ((size_t)((ct * KS + ks) * 32 + lane)) * 8;
    v8u bu;
#pragma unroll
    for (int v = 0; v < 8; ++v) bu[v] = p[v];
    return __builtin_bit_cast(v16bf, bu);
}

// ---------------------------------------------------------------------------
// Packing kernels
// ---------------------------------------------------------------------------
// Weights [K][256] fp32 -> fragment-order bf16 pairs:
// dst[((ct*KS + ks)*32 + lane)*8 + v] = {W[2*kp][col], W[2*kp+1][col]}
//   kp = ks*16 + (v&4)*2 + (lane>>4)*4 + (v&3) ; col = ct*16 + (lane&15); KS = K/32.
__global__ void pack_w_sw_k(const float* __restrict__ src,
                            unsigned* __restrict__ dst, int KS, int C) {
    int i = blockIdx.x * blockDim.x + threadIdx.x;
    int total = KS * 16 * 32 * 8;
    if (i >= total) return;
    int v    = i & 7;
    int lane = (i >> 3) & 31;
    int blk  = i >> 8;             // ct*KS + ks
    int ks = blk % KS, ct = blk / KS;
    int n = lane & 15, g = lane >> 4;
    int kp  = ks * 16 + (v & 4) * 2 + g * 4 + (v & 3);
    int col = ct * 16 + n;
    dst[i] = pack2(src[(size_t)(2 * kp) * C + col],
                   src[(size_t)(2 * kp + 1) * C + col]);
}

// Activations: pack along contiguous columns (K dim): one u32 per float2
__global__ void pack_cols_k(const float* __restrict__ src,
                            unsigned* __restrict__ dst, int total_u32) {
    int i = blockIdx.x * blockDim.x + threadIdx.x;
    if (i >= total_u32) return;
    float2 f = ((const float2*)src)[i];
    dst[i] = pack2(f.x, f.y);
}

// ---------------------------------------------------------------------------
// Kernel 1: xf = x @ W_in2f   [4096,256] x [256,256] -> fp32
// One WG = 16 rows x 256 cols; 8 waves, 2 col-tiles each; K = 256 (8 steps).
// ---------------------------------------------------------------------------
__global__ __launch_bounds__(256) void xf_gemm_k(const unsigned* __restrict__ xp,
                                                 const unsigned* __restrict__ Wi2fp,
                                                 float* __restrict__ xf) {
    __shared__ __align__(16) unsigned sA[16 * 128];   // 16 rows x 128 k-pairs
    const int rowbase = blockIdx.x * 16;
    for (int e = threadIdx.x; e < 16 * 128; e += 256)
        sA[e] = xp[(size_t)rowbase * 128 + e];
    __syncthreads();

    const int wv = threadIdx.x >> 5, lane = threadIdx.x & 31;
    const int n = lane & 15, g = lane >> 4;
#pragma unroll
    for (int t = 0; t < 2; ++t) {
        const int ct = wv * 2 + t;
        v8f acc = {};
#pragma unroll
        for (int ks = 0; ks < 8; ++ks) {
            v16bf a = load_afrag(sA, 128, 0, ks * 16, lane);
            v16bf b = load_bfrag_sw(Wi2fp, 8, ks, ct, lane);
            acc = wmma_bf16(a, b, acc);
        }
#pragma unroll
        for (int v = 0; v < 8; ++v)
            xf[(size_t)(rowbase + v + 8 * g) * 256 + ct * 16 + n] = acc[v];
    }
}

// ---------------------------------------------------------------------------
// Kernel 2 (main fused): one WG per (b,a). Filter MLP (2 WMMA GEMMs + ssp),
// cutoff*mask row scaling, neighbor gather from xf, reduce over N=64 -> y,
// emit y as bf16 pairs for the tail GEMMs.
// ---------------------------------------------------------------------------
__global__ __launch_bounds__(256) void main_fused_k(
    const float* __restrict__ r_ij, const float* __restrict__ f_ij,
    const float* __restrict__ nmask, const int* __restrict__ neighbors,
    const unsigned* __restrict__ W1p, const unsigned* __restrict__ W2p,
    const float* __restrict__ b1, const float* __restrict__ b2,
    const float* __restrict__ xf, unsigned* __restrict__ Yp) {

    __shared__ __align__(16) unsigned sF[64 * 64];   // f_ij bf16 pairs, 16 KB
    __shared__ __align__(16) bf16     sH[64 * 256];  // hidden after ssp, 32 KB
    __shared__ float sScale[64];
    __shared__ int   sNb[64];
    __shared__ float sY[256];

    const int ba = blockIdx.x;            // b*512 + a
    const int b  = ba >> 9;

    // stage f_ij tile: float4 loads -> bf16 pair-packed u32 (uint2 DS stores)
    const float4* fij4 = (const float4*)(f_ij + (size_t)ba * 64 * 128);
    for (int e = threadIdx.x; e < 64 * 32; e += 256) {
        float4 q = fij4[e];
        uint2 two;
        two.x = pack2(q.x, q.y);
        two.y = pack2(q.z, q.w);
        ((uint2*)sF)[e] = two;
    }
    if (threadIdx.x < 64) {
        const int m = threadIdx.x;
        const float r = r_ij[(size_t)ba * 64 + m];
        sScale[m] = (r <= 5.0f ? 1.0f : 0.0f) * nmask[(size_t)ba * 64 + m];
        sNb[m]    = neighbors[(size_t)ba * 64 + m];
    }
    sY[threadIdx.x] = 0.0f;
    __syncthreads();

    const int wv = threadIdx.x >> 5, lane = threadIdx.x & 31;
    const int n = lane & 15, g = lane >> 4;
    const v8f vzero = {};

    // ---- GEMM1: H = ssp(F[64x128] @ W1[128x256] + b1) ----
#pragma unroll
    for (int t = 0; t < 2; ++t) {
        const int ct = wv * 2 + t;
        v8f acc[4];
#pragma unroll
        for (int rt = 0; rt < 4; ++rt) acc[rt] = vzero;
#pragma unroll
        for (int ks = 0; ks < 4; ++ks) {
            v16bf bfr = load_bfrag_sw(W1p, 4, ks, ct, lane);
#pragma unroll
            for (int rt = 0; rt < 4; ++rt) {
                v16bf a = load_afrag(sF, 64, rt * 16, ks * 16, lane);
                acc[rt] = wmma_bf16(a, bfr, acc[rt]);
            }
        }
        const float b1v = b1[ct * 16 + n];
#pragma unroll
        for (int rt = 0; rt < 4; ++rt)
#pragma unroll
            for (int v = 0; v < 8; ++v) {
                const int m = rt * 16 + v + 8 * g;
                sH[(size_t)m * 256 + ct * 16 + n] = (bf16)ssp(acc[rt][v] + b1v);
            }
    }
    __syncthreads();

    // ---- GEMM2: W = H[64x256] @ W2[256x256] + b2; scale, gather, reduce ----
    const float* xfb = xf + (size_t)b * 512 * 256;
#pragma unroll
    for (int t = 0; t < 2; ++t) {
        const int ct = wv * 2 + t;
        const int f  = ct * 16 + n;
        v8f acc[4];
#pragma unroll
        for (int rt = 0; rt < 4; ++rt) acc[rt] = vzero;
#pragma unroll
        for (int ks = 0; ks < 8; ++ks) {
            v16bf bfr = load_bfrag_sw(W2p, 8, ks, ct, lane);
#pragma unroll
            for (int rt = 0; rt < 4; ++rt) {
                v16bf a = load_afrag((const unsigned*)sH, 128, rt * 16, ks * 16, lane);
                acc[rt] = wmma_bf16(a, bfr, acc[rt]);
            }
        }
        const float b2v = b2[f];
        float partial = 0.0f;
#pragma unroll
        for (int rt = 0; rt < 4; ++rt)
#pragma unroll
            for (int v = 0; v < 8; ++v) {
                const int m = rt * 16 + v + 8 * g;                  // neighbor slot
                const float Wv = (acc[rt][v] + b2v) * sScale[m];
                const float xv = xfb[(size_t)sNb[m] * 256 + f];     // gather
                partial += Wv * xv;
            }
        atomicAdd(&sY[f], partial);   // ds_add_f32; 2 adds per f (g=0 and g=1)
    }
    __syncthreads();

    if (threadIdx.x < 128)
        Yp[(size_t)ba * 128 + threadIdx.x] =
            pack2(sY[2 * threadIdx.x], sY[2 * threadIdx.x + 1]);
}

// ---------------------------------------------------------------------------
// Kernel 3 (tail fused, 16 atoms/WG):
//   v1  = Y @ W_f2out + b_f2out          (bf16 -> LDS)
//   out = ssp(v1 @ W_dense + G @ W_ang + b_dense)
// Both tail GEMMs chain into the same f32 WMMA accumulator.
// ---------------------------------------------------------------------------
__global__ __launch_bounds__(256) void tail_fused_k(
    const unsigned* __restrict__ Yp, const unsigned* __restrict__ Gp,
    const unsigned* __restrict__ Wf2op, const unsigned* __restrict__ Wdnp,
    const unsigned* __restrict__ Wangp,
    const float* __restrict__ b_f2out, const float* __restrict__ b_dense,
    float* __restrict__ out) {

    __shared__ __align__(16) unsigned sYp[16 * 128];  // 8 KB
    __shared__ __align__(16) unsigned sGp[16 * 64];   // 4 KB
    __shared__ __align__(16) bf16     sV1[16 * 256];  // 8 KB

    const int rowbase = blockIdx.x * 16;
    for (int e = threadIdx.x; e < 16 * 128; e += 256) sYp[e] = Yp[(size_t)rowbase * 128 + e];
    for (int e = threadIdx.x; e < 16 * 64;  e += 256) sGp[e] = Gp[(size_t)rowbase * 64 + e];
    __syncthreads();

    const int wv = threadIdx.x >> 5, lane = threadIdx.x & 31;
    const int n = lane & 15, g = lane >> 4;

#pragma unroll
    for (int t = 0; t < 2; ++t) {
        const int ct = wv * 2 + t;
        v8f acc = {};
#pragma unroll
        for (int ks = 0; ks < 8; ++ks) {
            v16bf a = load_afrag(sYp, 128, 0, ks * 16, lane);
            v16bf b = load_bfrag_sw(Wf2op, 8, ks, ct, lane);
            acc = wmma_bf16(a, b, acc);
        }
        const float bv = b_f2out[ct * 16 + n];
#pragma unroll
        for (int v = 0; v < 8; ++v)
            sV1[(size_t)(v + 8 * g) * 256 + ct * 16 + n] = (bf16)(acc[v] + bv);
    }
    __syncthreads();

#pragma unroll
    for (int t = 0; t < 2; ++t) {
        const int ct = wv * 2 + t;
        v8f acc = {};
#pragma unroll
        for (int ks = 0; ks < 8; ++ks) {             // v1 @ W_dense, K=256
            v16bf a = load_afrag((const unsigned*)sV1, 128, 0, ks * 16, lane);
            v16bf b = load_bfrag_sw(Wdnp, 8, ks, ct, lane);
            acc = wmma_bf16(a, b, acc);
        }
#pragma unroll
        for (int ks = 0; ks < 4; ++ks) {             // + G @ W_ang, K=128
            v16bf a = load_afrag(sGp, 64, 0, ks * 16, lane);
            v16bf b = load_bfrag_sw(Wangp, 4, ks, ct, lane);
            acc = wmma_bf16(a, b, acc);
        }
        const float bv = b_dense[ct * 16 + n];
#pragma unroll
        for (int v = 0; v < 8; ++v)
            out[(size_t)(rowbase + v + 8 * g) * 256 + ct * 16 + n] = ssp(acc[v] + bv);
    }
}

// ---------------------------------------------------------------------------
// Host launcher
// ---------------------------------------------------------------------------
extern "C" void kernel_launch(void* const* d_in, const int* in_sizes, int n_in,
                              void* d_out, int out_size, void* d_ws, size_t ws_size,
                              hipStream_t stream) {
    (void)in_sizes; (void)n_in; (void)out_size; (void)ws_size;
    const float* x      = (const float*)d_in[0];   // [8,512,256]
    const float* r_ij   = (const float*)d_in[1];   // [8,512,64]
    const float* f_ij   = (const float*)d_in[2];   // [8,512,64,128]
    const float* G_i    = (const float*)d_in[3];   // [8,512,128]
    const float* nmask  = (const float*)d_in[4];   // [8,512,64]
    const int*   nbrs   = (const int*)  d_in[5];   // [8,512,64]
    const float* W_in2f = (const float*)d_in[6];   // [256,256]
    const float* W1     = (const float*)d_in[7];   // [128,256]
    const float* b1     = (const float*)d_in[8];
    const float* W2     = (const float*)d_in[9];   // [256,256]
    const float* b2     = (const float*)d_in[10];
    const float* Wf2o   = (const float*)d_in[11];  // [256,256]
    const float* bf2o   = (const float*)d_in[12];
    const float* Wdn    = (const float*)d_in[13];  // [256,256]
    const float* bdn    = (const float*)d_in[14];
    const float* Wang   = (const float*)d_in[15];  // [128,256]
    float* out = (float*)d_out;

    char* ws = (char*)d_ws;
    unsigned* xp    = (unsigned*)(ws + 0);          // 4096*128 u32 = 2 MB
    unsigned* Gp    = (unsigned*)(ws + 2097152);    // 4096*64  u32 = 1 MB
    unsigned* Wi2fp = (unsigned*)(ws + 3145728);    // 128*256  u32
    unsigned* W1p   = (unsigned*)(ws + 3276800);    // 64*256   u32
    unsigned* W2p   = (unsigned*)(ws + 3342336);    // 128*256  u32
    unsigned* Wf2op = (unsigned*)(ws + 3473408);    // 128*256  u32
    unsigned* Wdnp  = (unsigned*)(ws + 3604480);    // 128*256  u32
    unsigned* Wangp = (unsigned*)(ws + 3735552);    // 64*256   u32
    float*    xf    = (float*)   (ws + 3801088);    // 4096*256 f32 = 4 MB
    unsigned* Yp    = (unsigned*)(ws + 7995392);    // 4096*128 u32 = 2 MB

    // Pack activations (pairs along contiguous K)
    pack_cols_k<<<2048, 256, 0, stream>>>(x,   xp, 4096 * 128);
    pack_cols_k<<<1024, 256, 0, stream>>>(G_i, Gp, 4096 * 64);
    // Pack weights into WMMA B-fragment order (KS = K/32)
    pack_w_sw_k<<<128, 256, 0, stream>>>(W_in2f, Wi2fp, 8, 256);
    pack_w_sw_k<<< 64, 256, 0, stream>>>(W1,     W1p,   4, 256);
    pack_w_sw_k<<<128, 256, 0, stream>>>(W2,     W2p,   8, 256);
    pack_w_sw_k<<<128, 256, 0, stream>>>(Wf2o,   Wf2op, 8, 256);
    pack_w_sw_k<<<128, 256, 0, stream>>>(Wdn,    Wdnp,  8, 256);
    pack_w_sw_k<<< 64, 256, 0, stream>>>(Wang,   Wangp, 4, 256);

    xf_gemm_k<<<256, 256, 0, stream>>>(xp, Wi2fp, xf);
    main_fused_k<<<4096, 256, 0, stream>>>(r_ij, f_ij, nmask, nbrs,
                                           W1p, W2p, b1, b2, xf, Yp);
    tail_fused_k<<<256, 256, 0, stream>>>(Yp, Gp, Wf2op, Wdnp, Wangp,
                                          bf2o, bdn, out);
}
